// ProductionMambaBlock_33809982554129
// MI455X (gfx1250) — matrix operand order repaired and checked by reading, not compiled, not run on hardware
//
#include <hip/hip_runtime.h>
#include <stdint.h>
#include <stddef.h>

#define D_MODEL 1024
#define D_STATE 16
#define D_INNER 2048
#define D_CONV  4
#define BATCH   2
#define SEQ     2048
#define NTOK    (BATCH * SEQ) /* 4096 */
#define LN_EPS  1e-5f

typedef __attribute__((ext_vector_type(16))) __bf16 v16bf;
typedef __attribute__((ext_vector_type(8)))  float  v8f;

__device__ __forceinline__ uint16_t f2bf(float f) {
    uint32_t u = __float_as_uint(f);
    u += 0x7FFFu + ((u >> 16) & 1u);   // round-to-nearest-even
    return (uint16_t)(u >> 16);
}
__device__ __forceinline__ float silu_f(float v) { return v / (1.f + __expf(-v)); }

// ---------------------------------------------------------------- fp32 -> bf16
__global__ __launch_bounds__(256) void convert_bf16(const float* __restrict__ src,
                                                    uint16_t* __restrict__ dst, int n) {
    int i = blockIdx.x * 256 + threadIdx.x;
    if (i < n) dst[i] = f2bf(src[i]);
}

// ------------------------------------------------------- LayerNorm -> bf16 row
__global__ __launch_bounds__(256) void layernorm_bf16(const float* __restrict__ x,
                                                      const float* __restrict__ w,
                                                      const float* __restrict__ b,
                                                      uint16_t* __restrict__ out) {
    __shared__ float s1[256], s2[256];
    const int tok = blockIdx.x;
    const float* row = x + (size_t)tok * D_MODEL;
    float v[4], sum = 0.f, sq = 0.f;
#pragma unroll
    for (int i = 0; i < 4; ++i) {
        v[i] = row[threadIdx.x + 256 * i];
        sum += v[i]; sq += v[i] * v[i];
    }
    s1[threadIdx.x] = sum; s2[threadIdx.x] = sq;
    __syncthreads();
    for (int off = 128; off > 0; off >>= 1) {
        if (threadIdx.x < off) {
            s1[threadIdx.x] += s1[threadIdx.x + off];
            s2[threadIdx.x] += s2[threadIdx.x + off];
        }
        __syncthreads();
    }
    const float mu = s1[0] * (1.f / D_MODEL);
    const float var = s2[0] * (1.f / D_MODEL) - mu * mu;
    const float rs = rsqrtf(var + LN_EPS);
#pragma unroll
    for (int i = 0; i < 4; ++i) {
        int j = threadIdx.x + 256 * i;
        out[(size_t)tok * D_MODEL + j] = f2bf((v[i] - mu) * rs * w[j] + b[j]);
    }
}

// --------------------------------------------------------------- WMMA bf16 GEMM
// C[M x N] (fp32, row-major, + optional fp32 residual) = A[M x K] @ W[N x K]^T
// Register-blocked: each wave computes a (16*BM) x (16*BN) tile.
// Per K-step (32): BM A-fragments + BN streamed B-fragments feed BM*BN WMMAs.
// Prefetch is UNconditional (branch-free inner loop); overshoot past the row
// end lands in valid adjacent workspace memory / is speculative-dropped.
template <int BM, int BN>
__global__ __launch_bounds__(256) void gemm_bf16_wmma(const uint16_t* __restrict__ A,
                                                      const uint16_t* __restrict__ W,
                                                      float* __restrict__ C,
                                                      const float* __restrict__ resid,
                                                      int M, int N, int K) {
    const int lane   = threadIdx.x & 31;
    const int wave   = threadIdx.x >> 5;
    const int tilesN = N / (16 * BN);
    const int tile   = blockIdx.x * 8 + wave;
    const int tm = tile / tilesN, tn = tile % tilesN;
    const int half = lane >> 4;       // 0: lanes 0-15, 1: lanes 16-31
    const int l16  = lane & 15;

    const uint16_t* Arow[BM];
    const uint16_t* Wrow[BN];
#pragma unroll
    for (int i = 0; i < BM; ++i)
        Arow[i] = A + (size_t)((tm * BM + i) * 16 + l16) * K;
#pragma unroll
    for (int j = 0; j < BN; ++j)
        Wrow[j] = W + (size_t)((tn * BN + j) * 16 + l16) * K;

    union Frag { v16bf v; uint32_t u[8]; };
    v8f acc[BM][BN];
#pragma unroll
    for (int i = 0; i < BM; ++i)
#pragma unroll
        for (int j = 0; j < BN; ++j) { v8f z = {}; acc[i][j] = z; }

    for (int kk = 0; kk < K; kk += 32) {
        Frag af[BM];
#pragma unroll
        for (int i = 0; i < BM; ++i) {
#pragma unroll
            for (int p = 0; p < 8; ++p) {
                // A 16-bit layout: half 0 -> K pairs {0,2,4,6,16,18,20,22}; half 1 -> +8
                const int aoff = ((p & 3) << 1) + ((p >> 2) << 4) + (half << 3);
                af[i].u[p] = *(const uint32_t*)(Arow[i] + kk + aoff);
            }
            __builtin_prefetch(Arow[i] + kk + 128, 0, 3);  // 256B ahead, no guard
        }
#pragma unroll
        for (int j = 0; j < BN; ++j) {
            Frag bf;
#pragma unroll
            for (int p = 0; p < 8; ++p) {
                // B 16-bit layout: half 0 -> K pairs {0,2,4,6,16,18,20,22}; half 1 -> +8
                const int boff = ((p & 3) << 1) + ((p >> 2) << 4) + (half << 3);
                bf.u[p] = *(const uint32_t*)(Wrow[j] + kk + boff);
            }
            __builtin_prefetch(Wrow[j] + kk + 128, 0, 3);
#pragma unroll
            for (int i = 0; i < BM; ++i)
                acc[i][j] = __builtin_amdgcn_wmma_f32_16x16x32_bf16(
                    false, af[i].v, false, bf.v, (short)0, acc[i][j], false, false);
        }
    }

#pragma unroll
    for (int i = 0; i < BM; ++i)
#pragma unroll
        for (int j = 0; j < BN; ++j)
#pragma unroll
            for (int r = 0; r < 8; ++r) {
                const int row = (tm * BM + i) * 16 + r + half * 8;
                const int col = (tn * BN + j) * 16 + l16;
                const size_t idx = (size_t)row * N + col;
                float v = acc[i][j][r];
                if (resid) v += resid[idx];
                C[idx] = v;
            }
}

// ---------------------------------------------- causal depthwise conv + SiLU
__global__ __launch_bounds__(256) void conv_silu(const float* __restrict__ xr,
                                                 const float* __restrict__ cw,
                                                 const float* __restrict__ cb,
                                                 float* __restrict__ xf,
                                                 uint16_t* __restrict__ xb) {
    const int c   = blockIdx.x * 256 + threadIdx.x;   // channel
    const int tok = blockIdx.y;                       // b*SEQ + t
    const int t   = tok & (SEQ - 1);
    float s = cb[c];
#pragma unroll
    for (int j = 0; j < D_CONV; ++j) {
        const int tt = t - (D_CONV - 1) + j;
        if (tt >= 0)
            s += cw[c * D_CONV + j] *
                 xr[(size_t)(tok - (D_CONV - 1) + j) * (2 * D_INNER) + c];
    }
    const float v = silu_f(s);
    xf[(size_t)tok * D_INNER + c] = v;
    xb[(size_t)tok * D_INNER + c] = f2bf(v);
}

// --------------------------------------------------- fused selective scan
// thread = one (batch, channel): 16 states in regs; dt projection fused;
// B/C staged in LDS per 64-step tile; epilogue: +D*x, *silu(res), -> bf16.
#define SCAN_TILE 64
__global__ __launch_bounds__(256) void scan_kernel(const float* __restrict__ x,
                                                   const float* __restrict__ xr,
                                                   const float* __restrict__ BC,
                                                   const float* __restrict__ A_log,
                                                   const float* __restrict__ dtW,
                                                   const float* __restrict__ dtb,
                                                   const float* __restrict__ Dp,
                                                   uint16_t* __restrict__ ybf) {
    __shared__ float sB[SCAN_TILE][D_STATE];
    __shared__ float sC[SCAN_TILE][D_STATE];
    const int c = blockIdx.x * 256 + threadIdx.x;
    const int b = blockIdx.y;

    float A[D_STATE], Wd[D_STATE], h[D_STATE];
#pragma unroll
    for (int n = 0; n < D_STATE; ++n) {
        A[n]  = -__expf(A_log[(size_t)c * D_STATE + n]);
        Wd[n] = dtW[(size_t)c * D_STATE + n];
        h[n]  = 0.f;
    }
    const float bias = dtb[c], Dc = Dp[c];

    for (int t0 = 0; t0 < SEQ; t0 += SCAN_TILE) {
        for (int i = threadIdx.x; i < SCAN_TILE * 32; i += 256) {
            const int tt = i >> 5, n = i & 31;
            const float v = BC[(size_t)(b * SEQ + t0 + tt) * 32 + n];
            if (n < D_STATE) sB[tt][n] = v; else sC[tt][n - D_STATE] = v;
        }
        __syncthreads();
        for (int tt = 0; tt < SCAN_TILE; ++tt) {
            const size_t tok = (size_t)b * SEQ + t0 + tt;
            const float xv  = x[tok * D_INNER + c];
            const float res = xr[tok * (2 * D_INNER) + D_INNER + c];
            float draw = bias;
#pragma unroll
            for (int n = 0; n < D_STATE; ++n) draw += sB[tt][n] * Wd[n];
            const float dt = (draw > 20.f) ? draw : __logf(1.f + __expf(draw));
            float y = 0.f;
#pragma unroll
            for (int n = 0; n < D_STATE; ++n) {
                const float dA = __expf(dt * A[n]);
                h[n] = h[n] * dA + (dt * sB[tt][n]) * xv;
                y += h[n] * sC[tt][n];
            }
            y = (y + Dc * xv) * silu_f(res);
            ybf[tok * D_INNER + c] = f2bf(y);
        }
        __syncthreads();
    }
}

// ------------------------------------------------------------------- launcher
extern "C" void kernel_launch(void* const* d_in, const int* in_sizes, int n_in,
                              void* d_out, int out_size, void* d_ws, size_t ws_size,
                              hipStream_t stream) {
    (void)in_sizes; (void)n_in; (void)out_size; (void)ws_size;
    const float* hs     = (const float*)d_in[0];
    const float* norm_w = (const float*)d_in[1];
    const float* norm_b = (const float*)d_in[2];
    const float* W1     = (const float*)d_in[3];   // (2*D_INNER, D_MODEL)
    const float* conv_w = (const float*)d_in[4];   // (D_INNER, 1, D_CONV)
    const float* conv_b = (const float*)d_in[5];
    const float* Wx     = (const float*)d_in[6];   // (32, D_INNER)
    const float* Wdt    = (const float*)d_in[7];   // (D_INNER, D_STATE)
    const float* bdt    = (const float*)d_in[8];
    const float* A_log  = (const float*)d_in[9];
    const float* Dp     = (const float*)d_in[10];
    const float* Wout   = (const float*)d_in[11];  // (D_MODEL, D_INNER)
    float* out = (float*)d_out;

    char* p = (char*)d_ws;
    auto alloc = [&](size_t bytes) {
        char* r = p; p += (bytes + 255) & ~(size_t)255; return (void*)r;
    };
    uint16_t* W1b   = (uint16_t*)alloc(sizeof(uint16_t) * (size_t)(2 * D_INNER) * D_MODEL);
    uint16_t* Woutb = (uint16_t*)alloc(sizeof(uint16_t) * (size_t)D_MODEL * D_INNER);
    uint16_t* Wxb   = (uint16_t*)alloc(sizeof(uint16_t) * (size_t)32 * D_INNER);
    uint16_t* hsb   = (uint16_t*)alloc(sizeof(uint16_t) * (size_t)NTOK * D_MODEL);
    float*    xrf   = (float*)   alloc(sizeof(float)    * (size_t)NTOK * 2 * D_INNER);
    float*    xf    = (float*)   alloc(sizeof(float)    * (size_t)NTOK * D_INNER);
    uint16_t* xb    = (uint16_t*)alloc(sizeof(uint16_t) * (size_t)NTOK * D_INNER);
    float*    bc    = (float*)   alloc(sizeof(float)    * (size_t)NTOK * 32);
    uint16_t* yb    = (uint16_t*)alloc(sizeof(uint16_t) * (size_t)NTOK * D_INNER);

    // weights -> bf16
    convert_bf16<<<(2 * D_INNER * D_MODEL + 255) / 256, 256, 0, stream>>>(W1, W1b, 2 * D_INNER * D_MODEL);
    convert_bf16<<<(D_MODEL * D_INNER + 255) / 256, 256, 0, stream>>>(Wout, Woutb, D_MODEL * D_INNER);
    convert_bf16<<<(32 * D_INNER + 255) / 256, 256, 0, stream>>>(Wx, Wxb, 32 * D_INNER);

    // LayerNorm -> bf16
    layernorm_bf16<<<NTOK, 256, 0, stream>>>(hs, norm_w, norm_b, hsb);

    // in_proj: (4096 x 1024) @ (1024 x 4096) -> x_and_res fp32
    // wave tile 32x64 -> (4096/32)*(4096/64)/8 = 1024 blocks
    gemm_bf16_wmma<2, 4><<<(NTOK / 32) * ((2 * D_INNER) / 64) / 8, 256, 0, stream>>>(
        hsb, W1b, xrf, nullptr, NTOK, 2 * D_INNER, D_MODEL);

    // causal depthwise conv + SiLU
    conv_silu<<<dim3(D_INNER / 256, NTOK), 256, 0, stream>>>(xrf, conv_w, conv_b, xf, xb);

    // x_proj: (4096 x 2048) @ (2048 x 32) -> B,C fp32 ; wave tile 32x32 -> 16 blocks
    gemm_bf16_wmma<2, 2><<<(NTOK / 32) * (32 / 32) / 8, 256, 0, stream>>>(
        xb, Wxb, bc, nullptr, NTOK, 32, D_INNER);

    // fused dt-proj + softplus + selective scan + D*x + SiLU gate -> y bf16
    scan_kernel<<<dim3(D_INNER / 256, BATCH), 256, 0, stream>>>(
        xf, xrf, bc, A_log, Wdt, bdt, Dp, yb);

    // out_proj with fused residual: (4096 x 2048) @ (2048 x 1024) + hs -> d_out
    // wave tile 32x64 -> (4096/32)*(1024/64)/8 = 256 blocks
    gemm_bf16_wmma<2, 4><<<(NTOK / 32) * (D_MODEL / 64) / 8, 256, 0, stream>>>(
        yb, Woutb, out, hs, NTOK, D_MODEL, D_INNER);
}
